// WideResNet_HyperDecisioNet_1_split_61349312856172
// MI455X (gfx1250) — matrix operand order repaired and checked by reading, not compiled, not run on hardware
//
#include <hip/hip_runtime.h>
#include <hip/hip_bf16.h>

// ============================================================================
// WideResNet / HyperDecisioNet forward on MI455X (gfx1250).
//
// All convs run as implicit-GEMM on v_wmma_f32_16x16x32_f16:
//   D(16 couts x 16 pos, f32) += A(16 couts x 32 K, f16) * B(32 K x 16 pos, f16)
// - Activations: fp16 NHWC -> each lane's 16-element K-slice is one contiguous
//   32B load (Cin in {16,64,96,192} keeps every half-slice inside one (r,s)).
// - Weights: device-packed into the A-fragment order used here
//   (contiguous K per lane half, matching B) and DMA'd to LDS with GLOBAL_LOAD_ASYNC_TO_LDS_B128
//   (ASYNCcnt / s_wait_asynccnt path, no VGPR round-trip).
// - Each wave computes 2 position-tiles per A fragment (2 WMMAs per ds_load).
// - Hyper-network rank-1 weights (w = h*W + B) folded into two shared-weight
//   convs + per-sample axpy: no 256-group conv anywhere.
// ============================================================================

typedef __attribute__((ext_vector_type(16))) _Float16 v16h;
typedef __attribute__((ext_vector_type(8)))  float    v8f;
typedef __attribute__((ext_vector_type(4)))  float    v4f;

static __device__ inline v8f zero8f() {
  v8f z;
#pragma unroll
  for (int i = 0; i < 8; ++i) z[i] = 0.f;
  return z;
}
static __device__ inline v16h zero16h() {
  v16h z;
#pragma unroll
  for (int i = 0; i < 16; ++i) z[i] = (_Float16)0.f;
  return z;
}

// ---------------------------------------------------------------------------
// Weight packing: OIHW f32 -> per-(coutTile, kstep, lane) A-fragment order.
// 16-bit A 16x32 (contiguous K, matching B): lane<16 (row m=lane) holds K = 0..15;
// lane>=16 (row m=lane-16) holds K = 16..31; v16h element e maps to
// kloc = base + e, base = (lane&16) ? 16 : 0.
// ---------------------------------------------------------------------------
__global__ void pack_w_k(const float* __restrict__ w, _Float16* __restrict__ dst,
                         int Cout, int Cin, int R, int KS, long long total) {
  long long t = (long long)blockIdx.x * 256 + threadIdx.x;
  if (t >= total) return;
  int e    = (int)(t & 15);
  int lane = (int)((t >> 4) & 31);
  long long q = t >> 9;
  int ks = (int)(q % KS);
  int ct = (int)(q / KS);
  int m    = lane & 15;
  int base = (lane & 16) ? 16 : 0;
  int kloc = base + e;
  int k    = ks * 32 + kloc;
  int K    = R * R * Cin;
  int cout = ct * 16 + m;
  float val = 0.f;
  if (k < K && cout < Cout) {
    int cin = k % Cin;
    int rs  = k / Cin;
    int r = rs / R, s = rs % R;  // R==1 -> r=s=0
    val = w[(((long long)cout * Cin + cin) * R + r) * R + s];
  }
  dst[t] = (_Float16)val;
}

// ---------------------------------------------------------------------------
// Implicit-GEMM conv. Wave = 16 couts x 32 positions (2 WMMA tiles / kstep).
// 8 waves/WG share one cout tile; A fragments async-DMA'd into LDS.
// B fragment (32x16): lane l = column l&15, K = ks*32 + (l<16?0:16) + e,
// contiguous -> one aligned 32B NHWC load per lane per tile per kstep.
// ---------------------------------------------------------------------------
template <int R, int STRIDE, bool ADD, int CIN>
__global__ __launch_bounds__(256) void conv_wmma(
    const _Float16* __restrict__ in, const _Float16* __restrict__ wpk,
    const float* __restrict__ res, float* __restrict__ out,
    int N, int H, int W, int Ho, int Wo, int Cout) {
  constexpr int K  = R * R * CIN;
  constexpr int KS = (K + 31) / 32;
  __shared__ _Float16 sA[KS * 512];  // <= 55296 B (K=1728)

  const int ct = blockIdx.y;
  {  // A-tile: global -> LDS via CDNA5 async DMA (no VGPR round-trip).
     // GV mode: vdst = LDS byte offset (sA is the only LDS object -> base 0),
     // vaddr = 64-bit global address. Tracked by ASYNCcnt.
    const _Float16* gsrc = wpk + (size_t)ct * (KS * 512);
    for (int i = threadIdx.x; i < KS * 64; i += 256) {
      unsigned lds = (unsigned)i * 16u;
      unsigned long long ga =
          (unsigned long long)(size_t)gsrc + (unsigned long long)i * 16u;
      asm volatile("global_load_async_to_lds_b128 %0, %1, off"
                   :: "v"(lds), "v"(ga) : "memory");
    }
    asm volatile("s_wait_asynccnt 0x0" ::: "memory");
  }
  __syncthreads();

  const int lane = threadIdx.x & 31;
  const int wave = threadIdx.x >> 5;
  const int col  = lane & 15;
  const long long pb = (long long)blockIdx.x * 256 + wave * 32 + col;
  const int HoWo = Ho * Wo;
  int n0, ho0, wo0, n1, ho1, wo1;
  { long long p = pb;      n0 = (int)(p / HoWo); int hw = (int)(p % HoWo); ho0 = hw / Wo; wo0 = hw % Wo; }
  { long long p = pb + 16; n1 = (int)(p / HoWo); int hw = (int)(p % HoWo); ho1 = hw / Wo; wo1 = hw % Wo; }
  const int khalf = (lane & 16) ? 16 : 0;

  // warm the B stream (speculative, emits global_prefetch_b8)
  __builtin_prefetch(in + (size_t)pb * CIN, 0, 1);

  v8f acc0 = zero8f(), acc1 = zero8f();
  for (int ks = 0; ks < KS; ++ks) {
    v16h a = *(const v16h*)(sA + ((ks * 32 + lane) << 4));  // 32B ds_load
    const int k0 = ks * 32 + khalf;
    v16h b0 = zero16h(), b1 = zero16h();
    if (k0 < K) {                       // only false for K%32==16 tail half
      const int rs = k0 / CIN, cin0 = k0 % CIN;  // compile-time strength-reduced
      const int r = (R == 3) ? (rs / 3) : 0;
      const int s = (R == 3) ? (rs % 3) : 0;
      {
        int hi = ho0 * STRIDE + r - (R >> 1), wi = wo0 * STRIDE + s - (R >> 1);
        if ((unsigned)hi < (unsigned)H && (unsigned)wi < (unsigned)W)
          b0 = *(const v16h*)(in + ((((long long)n0 * H + hi) * W + wi) * CIN + cin0));
      }
      {
        int hi = ho1 * STRIDE + r - (R >> 1), wi = wo1 * STRIDE + s - (R >> 1);
        if ((unsigned)hi < (unsigned)H && (unsigned)wi < (unsigned)W)
          b1 = *(const v16h*)(in + ((((long long)n1 * H + hi) * W + wi) * CIN + cin0));
      }
    }
    acc0 = __builtin_amdgcn_wmma_f32_16x16x32_f16(false, a, false, b0,
                                                  (short)0, acc0, false, false);
    acc1 = __builtin_amdgcn_wmma_f32_16x16x32_f16(false, a, false, b1,
                                                  (short)0, acc1, false, false);
  }

  // C/D layout: lane<16 -> rows m=0..7 of column n=lane; lane>=16 -> m=8..15.
  const int cbase = ct * 16 + ((lane & 16) ? 8 : 0);
  {
    long long ob = pb * (long long)Cout + cbase;
    if (ADD) {
      const v4f* rp = (const v4f*)(res + ob);
      v4f r0 = rp[0], r1 = rp[1];
#pragma unroll
      for (int j = 0; j < 4; ++j) { acc0[j] += r0[j]; acc0[j + 4] += r1[j]; }
    }
    v4f o0 = {acc0[0], acc0[1], acc0[2], acc0[3]};
    v4f o1 = {acc0[4], acc0[5], acc0[6], acc0[7]};
    *(v4f*)(out + ob) = o0;
    *(v4f*)(out + ob + 4) = o1;
  }
  {
    long long ob = (pb + 16) * (long long)Cout + cbase;
    if (ADD) {
      const v4f* rp = (const v4f*)(res + ob);
      v4f r0 = rp[0], r1 = rp[1];
#pragma unroll
      for (int j = 0; j < 4; ++j) { acc1[j] += r0[j]; acc1[j + 4] += r1[j]; }
    }
    v4f o0 = {acc1[0], acc1[1], acc1[2], acc1[3]};
    v4f o1 = {acc1[4], acc1[5], acc1[6], acc1[7]};
    *(v4f*)(out + ob) = o0;
    *(v4f*)(out + ob + 4) = o1;
  }
}

// ---------------------------------------------------------------------------
// Small helper kernels
// ---------------------------------------------------------------------------
__global__ void bn_coeff_k(const float* g, const float* b, const float* m,
                           const float* v, float* scale, float* shift, int C) {
  int c = threadIdx.x;
  if (c < C) {
    float s = g[c] * rsqrtf(v[c] + 1e-5f);
    scale[c] = s;
    shift[c] = b[c] - m[c] * s;
  }
}

// mode=1: fp16(relu(x*scale+shift)); mode=0: plain fp32->fp16 pack
__global__ void bn_relu_pack_k(const float* __restrict__ in,
                               const float* __restrict__ scale,
                               const float* __restrict__ shift,
                               _Float16* __restrict__ out,
                               long long total, int C, int mode) {
  long long i = (long long)blockIdx.x * 256 + threadIdx.x;
  if (i >= total) return;
  float v = in[i];
  if (mode) {
    int c = (int)(i % C);
    v = fmaxf(fmaf(v, scale[c], shift[c]), 0.f);
  }
  out[i] = (_Float16)v;
}

// conv1: 3->16 ch, 3x3 pad1. x NCHW f32 -> out NHWC f32 (C=16). Tiny.
__global__ void conv1_k(const float* __restrict__ x, const float* __restrict__ w,
                        float* __restrict__ out, long long total) {
  long long i = (long long)blockIdx.x * 256 + threadIdx.x;
  if (i >= total) return;
  int co = (int)(i & 15);
  int wo_ = (int)((i >> 4) & 31);
  int ho  = (int)((i >> 9) & 31);
  int n   = (int)(i >> 14);
  float acc = 0.f;
  for (int ci = 0; ci < 3; ++ci)
    for (int r = 0; r < 3; ++r) {
      int hi = ho + r - 1;
      if ((unsigned)hi >= 32u) continue;
      for (int s = 0; s < 3; ++s) {
        int wi = wo_ + s - 1;
        if ((unsigned)wi >= 32u) continue;
        acc = fmaf(x[((n * 3 + ci) * 32 + hi) * 32 + wi],
                   w[((co * 3 + ci) * 3 + r) * 3 + s], acc);
      }
    }
  out[i] = acc;
}

// global average pool over P positions (NHWC); optional fused bn+relu
__global__ void gap_k(const float* __restrict__ in, const float* __restrict__ scale,
                      const float* __restrict__ shift, float* __restrict__ f,
                      int P, int C) {
  int n = blockIdx.x, c = threadIdx.x;
  if (c >= C) return;
  const float* base = in + (long long)n * P * C + c;
  float acc = 0.f;
  for (int p = 0; p < P; ++p) {
    float v = base[(long long)p * C];
    if (scale) v = fmaxf(fmaf(v, scale[c], shift[c]), 0.f);
    acc += v;
  }
  f[n * C + c] = acc / (float)P;
}

// 2-way selector: softmax over f@selw.T+selb; writes sr/sb/sbr + h scalars
__global__ void select_k(const float* __restrict__ f, const float* __restrict__ selw,
                         const float* __restrict__ selb, float* __restrict__ out,
                         float* __restrict__ h0, float* __restrict__ h1,
                         int N, int C, int offSB, int offSR, int offSBR) {
  int n = blockIdx.x * blockDim.x + threadIdx.x;
  if (n >= N) return;
  float l0 = selb[0], l1 = selb[1];
  for (int c = 0; c < C; ++c) {
    float x = f[n * C + c];
    l0 = fmaf(x, selw[c], l0);
    l1 = fmaf(x, selw[C + c], l1);
  }
  float mx = fmaxf(l0, l1);
  float e0 = expf(l0 - mx), e1 = expf(l1 - mx);
  float inv = 1.f / (e0 + e1);
  float r0 = e0 * inv, r1 = e1 * inv;
  out[offSR + n * 2 + 0] = r0;
  out[offSR + n * 2 + 1] = r1;
  out[offSB + n] = (r1 > r0) ? 1.f : 0.f;
  out[offSBR + n] = r1;
  // h = (sr > 1e-4)*INITIAL_SIGMA + sr/SCALE_FACTOR
  h0[n] = ((r0 > 1e-4f) ? 0.5f : 0.f) + r0 * (1.f / 64.f);
  h1[n] = ((r1 > 1e-4f) ? 0.5f : 0.f) + r1 * (1.f / 64.f);
}

// y = h[n]*cw + cb (+res)  — rank-1 hypernet weight fold
__global__ void hyper_combine_k(const float* __restrict__ cw, const float* __restrict__ cb,
                                const float* __restrict__ h, const float* __restrict__ res,
                                float* __restrict__ out, long long perS, long long total) {
  long long i = (long long)blockIdx.x * 256 + threadIdx.x;
  if (i >= total) return;
  int n = (int)(i / perS);
  float v = fmaf(h[n], cw[i], cb[i]);
  if (res) v += res[i];
  out[i] = v;
}

__global__ void merge_k(const float* __restrict__ x0, const float* __restrict__ x1,
                        const float* __restrict__ sb, float* __restrict__ out,
                        long long perS, long long total) {
  long long i = (long long)blockIdx.x * 256 + threadIdx.x;
  if (i >= total) return;
  int n = (int)(i / perS);
  float s = sb[n];
  out[i] = (1.f - s) * x0[i] + s * x1[i];
}

__global__ void fc_k(const float* __restrict__ f, const float* __restrict__ w,
                     const float* __restrict__ b, float* __restrict__ out,
                     int C, int K) {
  int n = blockIdx.x, k = threadIdx.x;
  if (k >= K) return;
  float acc = b[k];
  for (int c = 0; c < C; ++c) acc = fmaf(f[n * C + c], w[k * C + c], acc);
  out[n * K + k] = acc;
}

// ---------------------------------------------------------------------------
// Host-side conv dispatcher (CIN is a template param -> compile-time K/KS)
// ---------------------------------------------------------------------------
template <int R, int ST, int CC>
static inline void conv_go(hipStream_t st, const _Float16* in, const _Float16* wpk,
                           const float* res, float* out,
                           int N, int H, int W, int Cout) {
  constexpr int pad = (R == 3) ? 1 : 0;
  int Ho = (H + 2 * pad - R) / ST + 1;
  int Wo = (W + 2 * pad - R) / ST + 1;
  long long P = (long long)N * Ho * Wo;  // always a multiple of 256 here
  dim3 g((unsigned)(P / 256), (unsigned)(Cout / 16));
  if (res) conv_wmma<R, ST, true , CC><<<g, 256, 0, st>>>(in, wpk, res,     out, N, H, W, Ho, Wo, Cout);
  else     conv_wmma<R, ST, false, CC><<<g, 256, 0, st>>>(in, wpk, nullptr, out, N, H, W, Ho, Wo, Cout);
}

static void conv_launch(hipStream_t st, int R, int stride,
                        const _Float16* in, const _Float16* wpk,
                        const float* res, float* out,
                        int N, int H, int W, int Cin, int Cout) {
#define GO(RR, SS, CC) \
  if (R == RR && stride == SS && Cin == CC) { conv_go<RR, SS, CC>(st, in, wpk, res, out, N, H, W, Cout); return; }
  GO(3, 1, 16) GO(3, 1, 64) GO(3, 1, 96) GO(3, 1, 192)
  GO(3, 2, 64) GO(3, 2, 96)
  GO(1, 1, 16) GO(1, 2, 64) GO(1, 2, 96)
#undef GO
}

// ============================================================================
// kernel_launch: full network. Input indices follow jax pytree order
// (top dict {'params','x'}: params first; dict keys sorted; lists in order).
// ============================================================================
extern "C" void kernel_launch(void* const* d_in, const int* in_sizes, int n_in,
                              void* d_out, int out_size, void* d_ws, size_t ws_size,
                              hipStream_t stream) {
  (void)in_sizes; (void)n_in; (void)out_size; (void)ws_size;
  auto F = [&](int i) -> const float* { return (const float*)d_in[i]; };
  const int N = 256;

  // ---- d_ws arena (~312 MB) ----
  char* ws = (char*)d_ws;
  const size_t MB = 1024ull * 1024ull;
  float*    F0  = (float*)(ws);               // 68 MB act region
  float*    F1  = (float*)(ws + 68  * MB);    // 68 MB
  float*    F2  = (float*)(ws + 136 * MB);    // 68 MB
  float*    F3  = (float*)(ws + 204 * MB);    // 26 MB
  _Float16* H0  = (_Float16*)(ws + 230 * MB); // 36 MB fp16 pack
  _Float16* H1  = (_Float16*)(ws + 266 * MB); // 36 MB fp16 pack
  _Float16* WPK = (_Float16*)(ws + 302 * MB); // 8 MB packed weights
  float*    SM  = (float*)(ws + 310 * MB);    // small scratch
  const long long HALF = 8ll * 1024 * 1024;   // floats: 32MB sub-slot offset

  float* bnScale = SM;
  float* bnShift = SM + 256;
  float* feat    = SM + 512;                  // up to 256*192
  float* h0      = SM + 512 + 49152;
  float* h1      = h0 + 256;
  float* dout    = (float*)d_out;
  const int OFF_SB = 25600, OFF_SR = 25856, OFF_SBR = 26368;

  // ---- pack all conv weights into WMMA A-fragment order ----
  size_t wo = 0;
  auto packW = [&](int idx, int Cout, int Cin, int R) -> const _Float16* {
    int KS = (R * R * Cin + 31) / 32, CT = (Cout + 15) / 16;
    _Float16* dst = WPK + wo;
    long long tot = (long long)CT * KS * 512;
    pack_w_k<<<(unsigned)((tot + 255) / 256), 256, 0, stream>>>(
        (const float*)d_in[idx], dst, Cout, Cin, R, KS, tot);
    wo += (size_t)tot;
    return dst;
  };

  const _Float16 *w1p[4], *w2p[4];
  w1p[0] = packW(28, 64, 16, 3);  w2p[0] = packW(29, 64, 64, 3);
  const _Float16* wsp1 = packW(30, 64, 16, 1);
  { int a[3] = {39, 49, 59}, b[3] = {40, 50, 60};
    for (int i = 0; i < 3; ++i) { w1p[i+1] = packW(a[i], 64, 64, 3); w2p[i+1] = packW(b[i], 64, 64, 3); } }
  const _Float16* hW1 = packW(16, 96, 64, 3);   // h1w as OIHW(96,64,3,3)
  const _Float16* hB1 = packW(15, 96, 64, 3);   // h1b
  const _Float16* hW2 = packW(18, 96, 96, 3);   // h2w
  const _Float16* hB2 = packW(17, 96, 96, 3);   // h2b
  const _Float16* hWS = packW(19, 96, 64, 1);   // hyper shortcut 1x1
  const _Float16 *l2w1[3], *l2w2[3];
  { int a[3] = {69, 79, 89}, b[3] = {70, 80, 90};
    for (int i = 0; i < 3; ++i) { l2w1[i] = packW(a[i], 96, 96, 3); l2w2[i] = packW(b[i], 96, 96, 3); } }
  const _Float16 *l3w1[4], *l3w2[4];
  l3w1[0] = packW(99, 192, 96, 3); l3w2[0] = packW(100, 192, 192, 3);
  const _Float16* wsp3 = packW(101, 192, 96, 1);
  { int a[3] = {110, 120, 130}, b[3] = {111, 121, 131};
    for (int i = 0; i < 3; ++i) { l3w1[i+1] = packW(a[i], 192, 192, 3); l3w2[i+1] = packW(b[i], 192, 192, 3); } }

  // ---- bn+relu+pack helpers (stream order makes coeff-buffer reuse safe) ----
  auto bnPack = [&](const float* act, int bnB, long long tot, int C, _Float16* dst) {
    // bn dict sorted {b,g,m,v}: g=+1, b=+0, m=+2, v=+3
    bn_coeff_k<<<1, 256, 0, stream>>>(F(bnB + 1), F(bnB), F(bnB + 2), F(bnB + 3),
                                      bnScale, bnShift, C);
    bn_relu_pack_k<<<(unsigned)((tot + 255) / 256), 256, 0, stream>>>(
        act, bnScale, bnShift, dst, tot, C, 1);
  };
  auto rawPack = [&](const float* act, long long tot, _Float16* dst) {
    bn_relu_pack_k<<<(unsigned)((tot + 255) / 256), 256, 0, stream>>>(
        act, nullptr, nullptr, dst, tot, 1, 0);
  };

  // ---- pre-activation residual block ----
  auto runBlock = [&](int bn1, int bn2, const _Float16* w1, const _Float16* w2,
                      const _Float16* wsw, float* x, float* u, float* nx,
                      int H, int W, int Cin, int Cout, int stride) {
    int Ho = (stride == 2) ? H / 2 : H, Wo = (stride == 2) ? W / 2 : W;
    long long inTot  = (long long)N * H * W * Cin;
    long long outTot = (long long)N * Ho * Wo * Cout;
    bnPack(x, bn1, inTot, Cin, H0);                       // t = relu(bn1(x))
    conv_launch(stream, 3, stride, H0, w1, nullptr, u, N, H, W, Cin, Cout);
    bnPack(u, bn2, outTot, Cout, H1);                     // t2 = relu(bn2(u))
    const float* res = x;                                 // identity shortcut
    if (wsw) {                                            // projection shortcut
      rawPack(x, inTot, H0);                              // raw x (no bn) -> f16
      conv_launch(stream, 1, stride, H0, wsw, nullptr, u, N, H, W, Cin, Cout);
      res = u;                                            // u's f32 is dead; reuse as sc
    }
    conv_launch(stream, 3, 1, H1, w2, res, nx, N, Ho, Wo, Cout, Cout);  // fused +res
  };

  // ================= network =================
  // conv1 (3->16), direct — 0.23 GFLOP, not worth WMMA
  { long long tot = (long long)N * 32 * 32 * 16;
    conv1_k<<<(unsigned)((tot + 255) / 256), 256, 0, stream>>>(
        (const float*)d_in[134], F(4), F0, tot); }

  // layer1 (4 blocks, 32x32): F0 -> F2 -> F0 -> F2 -> F0
  runBlock(20, 24, w1p[0], w2p[0], wsp1,    F0, F1, F2, 32, 32, 16, 64, 1);
  runBlock(31, 35, w1p[1], w2p[1], nullptr, F2, F1, F0, 32, 32, 64, 64, 1);
  runBlock(41, 45, w1p[2], w2p[2], nullptr, F0, F1, F2, 32, 32, 64, 64, 1);
  runBlock(51, 55, w1p[3], w2p[3], nullptr, F2, F1, F0, 32, 32, 64, 64, 1);

  // selector
  gap_k<<<N, 64, 0, stream>>>(F0, nullptr, nullptr, feat, 32 * 32, 64);
  select_k<<<1, N, 0, stream>>>(feat, F(133), F(132), dout, h0, h1,
                                N, 64, OFF_SB, OFF_SR, OFF_SBR);

  // hyper block (rank-1 weight fold): shared convs + per-sample axpy
  long long xTot = (long long)N * 32 * 32 * 64;
  long long bTot = (long long)N * 16 * 16 * 96;
  long long perS = 16 * 16 * 96;
  unsigned cgrid = (unsigned)((bTot + 255) / 256);
  bnPack(F0, 7, xTot, 64, H0);                         // relu(bn1(x)), shared
  float* cw = F1; float* cb = F1 + HALF;
  conv_launch(stream, 3, 2, H0, hW1, nullptr, cw, N, 32, 32, 64, 96);
  conv_launch(stream, 3, 2, H0, hB1, nullptr, cb, N, 32, 32, 64, 96);
  rawPack(F0, xTot, H1);                               // raw x for shortcut
  float* scH = F2;
  conv_launch(stream, 1, 2, H1, hWS, nullptr, scH, N, 32, 32, 64, 96);
  float* y0 = F0; float* y1 = F0 + HALF;               // x f32 now dead
  hyper_combine_k<<<cgrid, 256, 0, stream>>>(cw, cb, h0, nullptr, y0, perS, bTot);
  hyper_combine_k<<<cgrid, 256, 0, stream>>>(cw, cb, h1, nullptr, y1, perS, bTot);
  // stage-2, branch 0
  float* cw2 = F1; float* cb2 = F1 + HALF;
  float* x0 = F3; float* x1 = F2 + HALF;
  bnPack(y0, 11, bTot, 96, H0);
  conv_launch(stream, 3, 1, H0, hW2, nullptr, cw2, N, 16, 16, 96, 96);
  conv_launch(stream, 3, 1, H0, hB2, nullptr, cb2, N, 16, 16, 96, 96);
  hyper_combine_k<<<cgrid, 256, 0, stream>>>(cw2, cb2, h0, scH, x0, perS, bTot);
  // stage-2, branch 1
  bnPack(y1, 11, bTot, 96, H0);
  conv_launch(stream, 3, 1, H0, hW2, nullptr, cw2, N, 16, 16, 96, 96);
  conv_launch(stream, 3, 1, H0, hB2, nullptr, cb2, N, 16, 16, 96, 96);
  hyper_combine_k<<<cgrid, 256, 0, stream>>>(cw2, cb2, h1, scH, x1, perS, bTot);

  // layer2 branch 0: F3 -> F0 -> F0+HALF -> F0
  runBlock(61, 65, l2w1[0], l2w2[0], nullptr, x0,        F1, F0,        16, 16, 96, 96, 1);
  runBlock(71, 75, l2w1[1], l2w2[1], nullptr, F0,        F1, F0 + HALF, 16, 16, 96, 96, 1);
  runBlock(81, 85, l2w1[2], l2w2[2], nullptr, F0 + HALF, F1, F0,        16, 16, 96, 96, 1);
  // layer2 branch 1: F2+HALF -> F3 -> F2 -> F3
  runBlock(61, 65, l2w1[0], l2w2[0], nullptr, x1, F1, F3, 16, 16, 96, 96, 1);
  runBlock(71, 75, l2w1[1], l2w2[1], nullptr, F3, F1, F2, 16, 16, 96, 96, 1);
  runBlock(81, 85, l2w1[2], l2w2[2], nullptr, F2, F1, F3, 16, 16, 96, 96, 1);

  // data-dependent merge: (1-sb)*x0 + sb*x1 -> F2
  merge_k<<<cgrid, 256, 0, stream>>>(F0, F3, dout + OFF_SB, F2, perS, bTot);

  // layer3: F2 -(s2)-> F0 -> F3 -> F0 -> F3
  runBlock(91,  95,  l3w1[0], l3w2[0], wsp3,    F2, F1, F0, 16, 16, 96,  192, 2);
  runBlock(102, 106, l3w1[1], l3w2[1], nullptr, F0, F1, F3, 8,  8,  192, 192, 1);
  runBlock(112, 116, l3w1[2], l3w2[2], nullptr, F3, F1, F0, 8,  8,  192, 192, 1);
  runBlock(122, 126, l3w1[3], l3w2[3], nullptr, F0, F1, F3, 8,  8,  192, 192, 1);

  // head: bn_f + relu fused into GAP, then FC -> logits
  bn_coeff_k<<<1, 256, 0, stream>>>(F(1), F(0), F(2), F(3), bnScale, bnShift, 192);
  gap_k<<<N, 192, 0, stream>>>(F3, bnScale, bnShift, feat, 64, 192);
  fc_k<<<N, 128, 0, stream>>>(feat, F(6), F(5), dout, 192, 100);
}
